// DeconvNet_NB_38543036514752
// MI455X (gfx1250) — matrix operand order repaired and checked by reading, not compiled, not run on hardware
//
#include <hip/hip_runtime.h>
#include <math.h>

// ---------------- problem constants ----------------
constexpr int NN    = 4096;   // nodes
constexpr int GG    = 2048;   // genes
constexpr int H1_   = 512;
constexpr int H2_   = 128;
constexpr int HEADS_= 8;
constexpr int CH_   = 64;     // H1/HEADS
constexpr int EMB_  = 16;
constexpr int C_    = 20;
constexpr int CIND_ = H2_ + EMB_;   // 144
constexpr int KPAD_ = 160;          // 144 padded to multiple of 32
constexpr float EPS_ = 1e-6f;

// ---------------- vector types for WMMA ----------------
typedef __bf16          v16bf __attribute__((ext_vector_type(16)));
typedef float           v8f   __attribute__((ext_vector_type(8)));
typedef unsigned short  us8   __attribute__((ext_vector_type(8)));
typedef unsigned short  us16  __attribute__((ext_vector_type(16)));

union Frag {
    us16  u;
    us8   h[2];
    v16bf v;
};

// ---------------- device helpers ----------------
__device__ __forceinline__ unsigned short f2bf(float f) {
    unsigned u = __float_as_uint(f);
    unsigned r = u + 0x7FFFu + ((u >> 16) & 1u);   // round-to-nearest-even
    return (unsigned short)(r >> 16);
}
__device__ __forceinline__ float sigm(float x)  { return 1.f / (1.f + expf(-x)); }
__device__ __forceinline__ float eluf(float x)  { return x > 0.f ? x : expm1f(x); }

__device__ __forceinline__ v8f wmma_bf16(const Frag& a, const Frag& b, v8f c) {
    return __builtin_amdgcn_wmma_f32_16x16x32_bf16(
               false, a.v, false, b.v, (short)0, c, false, false);
}

// ---------------- fp32 -> bf16 staging ----------------
__global__ void f32_to_bf16_k(const float* __restrict__ src,
                              unsigned short* __restrict__ dst, int n) {
    int i = blockIdx.x * blockDim.x + threadIdx.x;
    int stride = gridDim.x * blockDim.x;
    for (; i < n; i += stride) dst[i] = f2bf(src[i]);
}

// pad decoder weight [512,144] -> bf16 [512,160]
__global__ void pad_wd1_k(const float* __restrict__ W, unsigned short* __restrict__ out) {
    int i = blockIdx.x * blockDim.x + threadIdx.x;
    if (i >= 512 * KPAD_) return;
    int r = i / KPAD_, c = i % KPAD_;
    out[i] = (c < CIND_) ? f2bf(W[r * CIND_ + c]) : (unsigned short)0;
}

// ---- WMMA GEMM: C[M,Nout] = A[M,K](bf16) @ B[Nout,K](bf16)^T + bias ----
// 2x2 register blocking: one wave owns a 32x32 output tile (4 accumulators),
// 4 WMMAs per 8 x b128 loads. 4 waves/block cover a 32(M) x 128(N) strip.
__global__ void gemm_bt_wmma(const unsigned short* __restrict__ A,
                             const unsigned short* __restrict__ B,
                             const float* __restrict__ bias,
                             float* __restrict__ Cout,
                             int M, int Nout, int K) {
    const int wave = threadIdx.x >> 5;
    const int lane = threadIdx.x & 31;
    const int n0 = blockIdx.x * 128 + wave * 32;
    const int m0 = blockIdx.y * 32;
    if (n0 >= Nout || m0 >= M) return;           // uniform per wave -> EXEC stays full
    const int half = lane >> 4;
    const int ln   = lane & 15;

    const unsigned short* __restrict__ arow0 = A + (size_t)(m0 + ln) * K;
    const unsigned short* __restrict__ arow1 = A + (size_t)(m0 + 16 + ln) * K;
    const unsigned short* __restrict__ brow0 = B + (size_t)(n0 + ln) * K;
    const unsigned short* __restrict__ brow1 = B + (size_t)(n0 + 16 + ln) * K;

    v8f acc00 = {}, acc01 = {}, acc10 = {}, acc11 = {};
    for (int k0 = 0; k0 < K; k0 += 32) {
        if (k0 + 32 < K) {
            __builtin_prefetch(arow0 + k0 + 32, 0, 3);
            __builtin_prefetch(arow1 + k0 + 32, 0, 3);
            __builtin_prefetch(brow0 + k0 + 32, 0, 3);
            __builtin_prefetch(brow1 + k0 + 32, 0, 3);
        }
        Frag a0, a1, b0, b1;
        // A 16x32 bf16 fragment: lanes 0-15 K=[0..7,16..23], lanes 16-31 K=[8..15,24..31]
        a0.h[0] = *(const us8*)(arow0 + k0 + half * 8);
        a0.h[1] = *(const us8*)(arow0 + k0 + 16 + half * 8);
        a1.h[0] = *(const us8*)(arow1 + k0 + half * 8);
        a1.h[1] = *(const us8*)(arow1 + k0 + 16 + half * 8);
        // B 32x16 fragment from row-major [Nout,K]: lane holds column n, same K order as A
        b0.h[0] = *(const us8*)(brow0 + k0 + half * 8);
        b0.h[1] = *(const us8*)(brow0 + k0 + 16 + half * 8);
        b1.h[0] = *(const us8*)(brow1 + k0 + half * 8);
        b1.h[1] = *(const us8*)(brow1 + k0 + 16 + half * 8);
        acc00 = wmma_bf16(a0, b0, acc00);
        acc01 = wmma_bf16(a0, b1, acc01);
        acc10 = wmma_bf16(a1, b0, acc10);
        acc11 = wmma_bf16(a1, b1, acc11);
    }
    const float bv0 = bias ? bias[n0 + ln] : 0.f;
    const float bv1 = bias ? bias[n0 + 16 + ln] : 0.f;
    // D layout: elem r -> row +r+half*8, col +ln
#pragma unroll
    for (int r = 0; r < 8; ++r) {
        const size_t row0 = (size_t)(m0 + r + half * 8) * Nout;
        const size_t row1 = (size_t)(m0 + 16 + r + half * 8) * Nout;
        Cout[row0 + n0 + ln]      = acc00[r] + bv0;
        Cout[row0 + n0 + 16 + ln] = acc01[r] + bv1;
        Cout[row1 + n0 + ln]      = acc10[r] + bv0;
        Cout[row1 + n0 + 16 + ln] = acc11[r] + bv1;
    }
}

// ---------------- GAT attention scores a0[h][n]=h_row.v0, a1[h][n]=h_row.v1 ------
__global__ void gat_scores_k(const float* __restrict__ h,    // [N, 512]
                             const float* __restrict__ v0,   // [HEADS*CH]
                             const float* __restrict__ v1,
                             float* __restrict__ a0, float* __restrict__ a1) {
    int t = blockIdx.x * blockDim.x + threadIdx.x;
    if (t >= NN * HEADS_) return;
    int n = t >> 3, head = t & 7;
    const float* hr = h + (size_t)n * H1_ + head * CH_;
    const float* w0 = v0 + head * CH_;
    const float* w1 = v1 + head * CH_;
    float s0 = 0.f, s1 = 0.f;
#pragma unroll 8
    for (int c = 0; c < CH_; ++c) { float x = hr[c]; s0 += x * w0[c]; s1 += x * w1[c]; }
    a0[(size_t)head * NN + n] = s0;
    a1[(size_t)head * NN + n] = s1;
}

// ---- sparse GAT aggregation + ELU: out[n,:] = elu(P @ h); fp32 + bf16 outputs ----
__global__ void gat_aggregate_k(const float* __restrict__ adj,  // [N,N], 0/1
                                const float* __restrict__ h,    // [N,512]
                                const float* __restrict__ a0,   // [8,N]
                                const float* __restrict__ a1,   // [8,N]
                                float* __restrict__ out,        // [N,512] fp32
                                unsigned short* __restrict__ out_bf) { // [N,512] bf16
    const int n   = blockIdx.x;
    const int tid = threadIdx.x;                  // 256 threads
    __shared__ int   s_idx[512];
    __shared__ int   s_cnt;
    __shared__ float s_den[HEADS_];

    const int h0 = tid >> 6;
    const int h1 = (tid + 256) >> 6;
    const float a00 = a0[(size_t)h0 * NN + n];
    const float a01 = a0[(size_t)h1 * NN + n];
    const float a0d = (tid < HEADS_) ? a0[(size_t)tid * NN + n] : 0.f;

    float acc0 = 0.f, acc1 = 0.f, dacc = 0.f;
    const float* adjrow = adj + (size_t)n * NN;

    for (int base = 0; base < NN; base += 512) {
        if (tid == 0) s_cnt = 0;
        __syncthreads();
        for (int j = base + tid; j < base + 512; j += 256)
            if (adjrow[j] != 0.f) { int p = atomicAdd(&s_cnt, 1); s_idx[p] = j; }
        __syncthreads();
        const int cnt = s_cnt;
        for (int t = 0; t < cnt; ++t) {
            const int j = s_idx[t];
            const float w0 = expf(sigm(a00 + a1[(size_t)h0 * NN + j]) - 0.5f);
            const float w1 = expf(sigm(a01 + a1[(size_t)h1 * NN + j]) - 0.5f);
            acc0 += w0 * h[(size_t)j * H1_ + tid];
            acc1 += w1 * h[(size_t)j * H1_ + tid + 256];
            if (tid < HEADS_)
                dacc += expf(sigm(a0d + a1[(size_t)tid * NN + j]) - 0.5f);
        }
        __syncthreads();
    }
    if (tid < HEADS_) s_den[tid] = dacc;
    __syncthreads();
    float d0 = s_den[h0]; d0 = (d0 == 0.f) ? 1.f : d0;
    float d1 = s_den[h1]; d1 = (d1 == 0.f) ? 1.f : d1;
    const float o0 = eluf(acc0 / d0);
    const float o1 = eluf(acc1 / d1);
    out[(size_t)n * H1_ + tid]          = o0;
    out[(size_t)n * H1_ + tid + 256]    = o1;
    out_bf[(size_t)n * H1_ + tid]       = f2bf(o0);
    out_bf[(size_t)n * H1_ + tid + 256] = f2bf(o1);
}

// ---------------- beta = softmax(elu(Z) @ Wb^T + bb) ----------------
__global__ void beta_k(const float* __restrict__ Z, const float* __restrict__ Wb,
                       const float* __restrict__ bb, float* __restrict__ beta) {
    const int n = blockIdx.x, tid = threadIdx.x;   // 128 threads
    __shared__ float s_ez[H2_];
    __shared__ float s_logit[C_];
    __shared__ float s_red[2];
    s_ez[tid] = eluf(Z[(size_t)n * H2_ + tid]);
    __syncthreads();
    if (tid < C_) {
        float s = bb[tid];
        for (int k = 0; k < H2_; ++k) s += Wb[tid * H2_ + k] * s_ez[k];
        s_logit[tid] = s;
    }
    __syncthreads();
    if (tid == 0) {
        float mx = -1e30f;
        for (int c = 0; c < C_; ++c) mx = fmaxf(mx, s_logit[c]);
        float sum = 0.f;
        for (int c = 0; c < C_; ++c) sum += expf(s_logit[c] - mx);
        s_red[0] = mx; s_red[1] = sum;
    }
    __syncthreads();
    if (tid < C_) beta[(size_t)n * C_ + tid] = expf(s_logit[tid] - s_red[0]) / s_red[1];
}

// ---- build Xd=concat(Z,emb) as padded bf16 + alpha = elu(Xd)@Wa^T+ba ----
__global__ void xd_alpha_k(const float* __restrict__ Z, const float* __restrict__ embt,
                           const int* __restrict__ slice, const float* __restrict__ Wa,
                           const float* __restrict__ ba,
                           unsigned short* __restrict__ Xd_bf, float* __restrict__ alpha) {
    const int n = blockIdx.x, tid = threadIdx.x;   // 160 threads
    const int sl = slice[n];
    float x = 0.f;
    if (tid < H2_)        x = Z[(size_t)n * H2_ + tid];
    else if (tid < CIND_) x = embt[sl * EMB_ + (tid - H2_)];
    Xd_bf[(size_t)n * KPAD_ + tid] = (tid < CIND_) ? f2bf(x) : (unsigned short)0;

    __shared__ float s_r[KPAD_];
    s_r[tid] = (tid < CIND_) ? eluf(x) * Wa[tid] : 0.f;
    __syncthreads();
    for (int off = 128; off > 0; off >>= 1) {
        if (tid < off && tid + off < KPAD_) s_r[tid] += s_r[tid + off];
        __syncthreads();
    }
    if (tid == 0) alpha[n] = s_r[0] + ba[0];
}

// ---------------- per-row feature loss: sqrt(sum (nf - xr)^2) ----------------
__global__ void feat_row_k(const float* __restrict__ nf, const float* __restrict__ xr,
                           float* __restrict__ row_feat) {
    const int n = blockIdx.x, tid = threadIdx.x;   // 256 threads
    float s = 0.f;
    for (int g = tid; g < GG; g += 256) {
        float d = nf[(size_t)n * GG + g] - xr[(size_t)n * GG + g];
        s += d * d;
    }
    __shared__ float sr[256];
    sr[tid] = s; __syncthreads();
    for (int off = 128; off > 0; off >>= 1) {
        if (tid < off) sr[tid] += sr[tid + off];
        __syncthreads();
    }
    if (tid == 0) row_feat[n] = sqrtf(sr[0]);
}

// ---------------- per-row NB log-likelihood sum ----------------
__global__ void nb_row_k(const float* __restrict__ count, const float* __restrict__ basis,
                         const float* __restrict__ beta,  const float* __restrict__ alpha,
                         const float* __restrict__ gamma, const float* __restrict__ logtheta,
                         const float* __restrict__ lib,   const int* __restrict__ slice,
                         float* __restrict__ row_ll) {
    const int n = blockIdx.x, tid = threadIdx.x;   // 256 threads
    __shared__ float s_beta[C_];
    if (tid < C_) s_beta[tid] = beta[(size_t)n * C_ + tid];
    __syncthreads();
    const int   sl = slice[n];
    const float al = alpha[n], ls = lib[n];
    float s = 0.f;
    for (int g = tid; g < GG; g += 256) {
        float dot = 0.f;
#pragma unroll
        for (int c = 0; c < C_; ++c) dot += s_beta[c] * basis[(size_t)c * GG + g];
        const float log_lam = logf(dot + EPS_) + al + gamma[(size_t)sl * GG + g];
        const float lam = expf(log_lam);
        const float th  = expf(logtheta[(size_t)sl * GG + g]);
        const float mu  = ls * lam;
        const float cn  = count[(size_t)n * GG + g];
        const float ll  = lgammaf(cn + th + EPS_) - lgammaf(th + EPS_)
                        + th * logf(th + EPS_)  - th * logf(th + mu + EPS_)
                        + cn * logf(mu + EPS_)  - cn * logf(th + mu + EPS_);
        s += ll;
    }
    __shared__ float sr[256];
    sr[tid] = s; __syncthreads();
    for (int off = 128; off > 0; off >>= 1) {
        if (tid < off) sr[tid] += sr[tid + off];
        __syncthreads();
    }
    if (tid == 0) row_ll[n] = sr[0];
}

// ---------------- final scalar ----------------
__global__ void final_reduce_k(const float* __restrict__ row_ll,
                               const float* __restrict__ row_feat,
                               float* __restrict__ out) {
    const int tid = threadIdx.x;   // 256
    float sll = 0.f, sft = 0.f;
    for (int n = tid; n < NN; n += 256) { sll += row_ll[n]; sft += row_feat[n]; }
    __shared__ float a[256], b[256];
    a[tid] = sll; b[tid] = sft; __syncthreads();
    for (int off = 128; off > 0; off >>= 1) {
        if (tid < off) { a[tid] += a[tid + off]; b[tid] += b[tid + off]; }
        __syncthreads();
    }
    if (tid == 0) out[0] = -(a[0] / (float)NN) + 0.1f * (b[0] / (float)NN);
}

// =============================== host ===============================
extern "C" void kernel_launch(void* const* d_in, const int* in_sizes, int n_in,
                              void* d_out, int out_size, void* d_ws, size_t ws_size,
                              hipStream_t stream) {
    const float* adj      = (const float*)d_in[0];
    const float* nf       = (const float*)d_in[1];
    const float* cntm     = (const float*)d_in[2];
    const float* lib      = (const float*)d_in[3];
    const int*   slice    = (const int*)  d_in[4];
    const float* basis    = (const float*)d_in[5];
    const float* We1      = (const float*)d_in[6];   // [512,2048] flat
    const float* be1      = (const float*)d_in[7];   // [512]
    const float* v0e1     = (const float*)d_in[8];
    const float* v1e1     = (const float*)d_in[9];
    const float* W2       = (const float*)d_in[10];  // [128,512]
    const float* b2       = (const float*)d_in[11];
    const float* Wd1      = (const float*)d_in[12];  // [512,144] flat
    const float* bd1      = (const float*)d_in[13];
    const float* v0d1     = (const float*)d_in[14];
    const float* v1d1     = (const float*)d_in[15];
    const float* Wd2      = (const float*)d_in[16];  // [2048,512]
    const float* bd2      = (const float*)d_in[17];
    const float* Wa       = (const float*)d_in[18];  // [144]
    const float* ba       = (const float*)d_in[19];
    const float* Wb       = (const float*)d_in[20];  // [20,128]
    const float* bb       = (const float*)d_in[21];
    const float* gamma    = (const float*)d_in[22];  // [4,2048]
    const float* logtheta = (const float*)d_in[23];
    const float* embt     = (const float*)d_in[24];  // [4,16]

    // ---- workspace carve-up (256B aligned) ----
    char* wp = (char*)d_ws;
    auto carve = [&](size_t bytes) -> void* {
        void* r = (void*)wp;
        wp += (bytes + 255) & ~(size_t)255;
        return r;
    };
    unsigned short* nf_bf   = (unsigned short*)carve((size_t)NN * GG * 2);
    unsigned short* We1_bf  = (unsigned short*)carve((size_t)H1_ * GG * 2);
    float*          h_enc   = (float*)carve((size_t)NN * H1_ * 4);
    float*          a0e     = (float*)carve((size_t)HEADS_ * NN * 4);
    float*          a1e     = (float*)carve((size_t)HEADS_ * NN * 4);
    float*          H_act   = (float*)carve((size_t)NN * H1_ * 4);
    unsigned short* H_bf    = (unsigned short*)carve((size_t)NN * H1_ * 2);
    unsigned short* W2_bf   = (unsigned short*)carve((size_t)H2_ * H1_ * 2);
    float*          Z       = (float*)carve((size_t)NN * H2_ * 4);
    unsigned short* Xd_bf   = (unsigned short*)carve((size_t)NN * KPAD_ * 2);
    float*          alpha   = (float*)carve((size_t)NN * 4);
    float*          beta    = (float*)carve((size_t)NN * C_ * 4);
    unsigned short* Wd1_bf  = (unsigned short*)carve((size_t)H1_ * KPAD_ * 2);
    float*          h_dec   = (float*)carve((size_t)NN * H1_ * 4);
    float*          a0d     = (float*)carve((size_t)HEADS_ * NN * 4);
    float*          a1d     = (float*)carve((size_t)HEADS_ * NN * 4);
    float*          Hd_act  = (float*)carve((size_t)NN * H1_ * 4);
    unsigned short* Hd_bf   = (unsigned short*)carve((size_t)NN * H1_ * 2);
    unsigned short* Wd2_bf  = (unsigned short*)carve((size_t)GG * H1_ * 2);
    float*          X_rec   = (float*)carve((size_t)NN * GG * 4);
    float*          row_ft  = (float*)carve((size_t)NN * 4);
    float*          row_ll  = (float*)carve((size_t)NN * 4);

    const dim3 gblk(128);   // 4 waves / block, one 32x32 tile per wave

    // ---- encoder GAT layer 1 ----
    f32_to_bf16_k<<<4096, 256, 0, stream>>>(nf,  nf_bf,  NN * GG);
    f32_to_bf16_k<<<4096, 256, 0, stream>>>(We1, We1_bf, H1_ * GG);
    gemm_bt_wmma<<<dim3(H1_ / 128, NN / 32), gblk, 0, stream>>>(
        nf_bf, We1_bf, be1, h_enc, NN, H1_, GG);
    gat_scores_k<<<(NN * HEADS_ + 255) / 256, 256, 0, stream>>>(h_enc, v0e1, v1e1, a0e, a1e);
    gat_aggregate_k<<<NN, 256, 0, stream>>>(adj, h_enc, a0e, a1e, H_act, H_bf);

    // ---- Z = H @ W2^T + b2 ----
    f32_to_bf16_k<<<256, 256, 0, stream>>>(W2, W2_bf, H2_ * H1_);
    gemm_bt_wmma<<<dim3(1, NN / 32), gblk, 0, stream>>>(
        H_bf, W2_bf, b2, Z, NN, H2_, H1_);

    // ---- deconvolution heads: beta, alpha; decoder input Xd ----
    beta_k<<<NN, H2_, 0, stream>>>(Z, Wb, bb, beta);
    xd_alpha_k<<<NN, KPAD_, 0, stream>>>(Z, embt, slice, Wa, ba, Xd_bf, alpha);

    // ---- decoder GAT layer 1 ----
    pad_wd1_k<<<(H1_ * KPAD_ + 255) / 256, 256, 0, stream>>>(Wd1, Wd1_bf);
    gemm_bt_wmma<<<dim3(H1_ / 128, NN / 32), gblk, 0, stream>>>(
        Xd_bf, Wd1_bf, bd1, h_dec, NN, H1_, KPAD_);
    gat_scores_k<<<(NN * HEADS_ + 255) / 256, 256, 0, stream>>>(h_dec, v0d1, v1d1, a0d, a1d);
    gat_aggregate_k<<<NN, 256, 0, stream>>>(adj, h_dec, a0d, a1d, Hd_act, Hd_bf);

    // ---- X_recon = Hd @ Wd2^T + bd2 ----
    f32_to_bf16_k<<<2048, 256, 0, stream>>>(Wd2, Wd2_bf, GG * H1_);
    gemm_bt_wmma<<<dim3(GG / 128, NN / 32), gblk, 0, stream>>>(
        Hd_bf, Wd2_bf, bd2, X_rec, NN, GG, H1_);

    // ---- losses ----
    feat_row_k<<<NN, 256, 0, stream>>>(nf, X_rec, row_ft);
    nb_row_k<<<NN, 256, 0, stream>>>(cntm, basis, beta, alpha, gamma, logtheta,
                                     lib, slice, row_ll);
    final_reduce_k<<<1, 256, 0, stream>>>(row_ll, row_ft, (float*)d_out);
}
